// GRUD_56831007260757
// MI455X (gfx1250) — matrix operand order repaired and checked by reading, not compiled, not run on hardware
//
#include <hip/hip_runtime.h>
#include <math.h>

// Problem constants (match reference).
#define B_   512
#define T_   256
#define D_   256
#define H_   1024
#define BT   64            // batch-tile rows per workgroup (2 m-subtiles per wave)
#define HT   64            // hidden-tile cols per workgroup (1 n-subtile per wave)
#define KTOT 512           // effective K = [x_tilde | m] (xm slice folded into bias const)
#define KSTR (KTOT + 8)    // padded K stride (f16 elems), keeps 16B align, staggers banks
#define NT_  (H_ / 16)     // 64 column tiles of 16

typedef float    v8f  __attribute__((ext_vector_type(8)));
typedef _Float16 v16h __attribute__((ext_vector_type(16)));
typedef _Float16 v8h  __attribute__((ext_vector_type(8)));
typedef _Float16 v4h  __attribute__((ext_vector_type(4)));

union HFrag { v16h v; v8h h[2]; };

// ---------------------------------------------------------------------------
// Prep 1: pack Wz/Wh into f16 *B-fragment order* so each wave can pull its
// time-invariant weight fragments into VGPRs with fully coalesced loads.
// Fragment layout: idx = ((g*NT_ + nt)*16 + ks)*512 + lane*16 + e
//   col = nt*16 + (lane&15)
//   k   = ks*32 + (lane>>4)*16 + e        (order used here, same as A loads: lanes0-15 K lo16, 16-31 K hi16)
//   src col-in-W = k<256 ? k : 512+(k-256)  (x-slice | m-slice; xm slice folded out)
// (r-gate is dead code in the reference: it never affects the output.)
// ---------------------------------------------------------------------------
__global__ void grud_pack_weights(const float* __restrict__ Wz,
                                  const float* __restrict__ Wh,
                                  _Float16* __restrict__ Wf) {
  int idx = blockIdx.x * blockDim.x + threadIdx.x;
  if (idx >= 2 * NT_ * 16 * 512) return;
  int g    = idx >> 19;            // gate: 0=z, 1=h
  int rem  = idx & ((1 << 19) - 1);
  int nt   = rem >> 13;            // column tile 0..63
  int rem2 = rem & 8191;
  int ks   = rem2 >> 9;            // K-step 0..15
  int rem3 = rem2 & 511;
  int lane = rem3 >> 4;            // 0..31
  int e    = rem3 & 15;            // element within fragment
  int col  = nt * 16 + (lane & 15);
  int k    = ks * 32 + (lane >> 4) * 16 + e;
  int srcc = (k < D_) ? k : (2 * D_ + (k - D_));
  const float* W = g ? Wh : Wz;
  Wf[idx] = (_Float16)W[(size_t)col * (3 * D_) + srcc];
}

// ---------------------------------------------------------------------------
// Prep 2: fold the time/batch-invariant xm contribution + bias into one fp32
// constant per output column (exact fp32 for this slice).
// ---------------------------------------------------------------------------
__global__ void grud_bias_const(const float* __restrict__ Wz, const float* __restrict__ bz,
                                const float* __restrict__ Wh, const float* __restrict__ bh,
                                const float* __restrict__ xm,
                                float* __restrict__ cz, float* __restrict__ ch) {
  int col = blockIdx.x * blockDim.x + threadIdx.x;
  if (col >= H_) return;
  float sz = bz[col], sh = bh[col];
  const float* wzr = Wz + (size_t)col * (3 * D_) + D_;
  const float* whr = Wh + (size_t)col * (3 * D_) + D_;
  for (int d = 0; d < D_; ++d) {
    float x = xm[d];
    sz += wzr[d] * x;
    sh += whr[d] * x;
  }
  cz[col] = sz;
  ch[col] = sh;
}

// ---------------------------------------------------------------------------
// Main: 128 WGs x 256 threads. Each WG owns a 64x64 tile of h over all T steps.
// Weights live in VGPRs (32 v16h fragments/wave, loaded once). A-tile
// (x_tilde | m) is double-buffered in LDS; one barrier per timestep.
// Per wave per step: 64 v_wmma_f32_16x16x32_f16 fed by 64 ds_load_b128,
// with A-fragments software-pipelined one K-step ahead to hide LDS latency.
// ---------------------------------------------------------------------------
__global__ __launch_bounds__(256, 2)
void grud_main(const float* __restrict__ X, const float* __restrict__ M,
               const float* __restrict__ xm, const float* __restrict__ gamma,
               const _Float16* __restrict__ Wf,
               const float* __restrict__ cz_all, const float* __restrict__ ch_all,
               float* __restrict__ h_buf) {
  extern __shared__ char smem_raw[];
  _Float16* sA0 = (_Float16*)smem_raw;                 // BT*KSTR
  _Float16* sA1 = sA0 + BT * KSTR;                     // BT*KSTR
  float* sGamma = (float*)(sA1 + BT * KSTR);           // D_
  float* sXm    = sGamma + D_;                         // D_

  const int tid  = threadIdx.x;
  const int lane = tid & 31;
  const int wave = tid >> 5;
  const int b0   = blockIdx.x * BT;
  const int mp   = wave >> 2;                          // m-subtile pair 0..1
  const int ntglob = blockIdx.y * (HT / 16) + (wave & 3);  // global 16-col tile 0..63
  const int lcol = lane & 15;
  const int lhi  = lane >> 4;

  for (int i = tid; i < D_; i += 256) {
    sGamma[i] = gamma[i];
    sXm[i]    = xm[i];
  }

  // ---- one-time: pull weight B-fragments into VGPRs (coalesced) ----
  v16h fz[16], fh[16];
  {
    const _Float16* pz = Wf + ((size_t)ntglob * 16) * 512 + lane * 16;
    const _Float16* ph = Wf + ((size_t)(NT_ + ntglob) * 16) * 512 + lane * 16;
#pragma unroll
    for (int ks = 0; ks < 16; ++ks) {
      fz[ks] = *(const v16h*)(pz + (size_t)ks * 512);
      fh[ks] = *(const v16h*)(ph + (size_t)ks * 512);
    }
  }
  const float czv = cz_all[ntglob * 16 + lcol];
  const float chv = ch_all[ntglob * 16 + lcol];

  // staging map: thread -> (row 0..63, 64-wide d chunk)
  const int srow = tid >> 2;
  const int sd0  = (tid & 3) * 64;

  auto stage = [&](int t, _Float16* sbuf) {
    const float* xr = X + ((size_t)(b0 + srow) * T_ + t) * D_ + sd0;
    const float* mr = M + ((size_t)(b0 + srow) * T_ + t) * D_ + sd0;
    _Float16* arow = sbuf + (size_t)srow * KSTR + sd0;
#pragma unroll
    for (int j = 0; j < 64; j += 4) {
      float4 xv = *(const float4*)&xr[j];
      float4 mv = *(const float4*)&mr[j];
      float xs[4] = {xv.x, xv.y, xv.z, xv.w};
      float ms[4] = {mv.x, mv.y, mv.z, mv.w};
      v4h xt4, m4;
#pragma unroll
      for (int c = 0; c < 4; ++c) {
        int   d  = sd0 + j + c;
        float mu = sXm[d];
        float om = 1.0f - ms[c];
        float xh = ms[c] * xs[c] + om * mu;           // masked fill
        float g  = __expf(-sGamma[d] * om);           // input decay
        xt4[c] = (_Float16)(g * xh + (1.0f - g) * mu);
        m4[c]  = (_Float16)ms[c];
      }
      *(v4h*)&arow[j]      = xt4;                     // K 0..255  : x_tilde
      *(v4h*)&arow[D_ + j] = m4;                      // K 256..511: m
    }
    __builtin_prefetch(xr + D_, 0, 1);                // speculative: next step's line
    __builtin_prefetch(mr + D_, 0, 1);
  };

  v8f h0 = {}, h1 = {};                               // persistent h (2 m-subtiles)
  stage(0, sA0);
  __syncthreads();

  for (int t = 0; t < T_; ++t) {
    _Float16* cur = (t & 1) ? sA1 : sA0;
    _Float16* nxt = (t & 1) ? sA0 : sA1;
    const _Float16* aB0 = cur + (size_t)(mp * 32 + lcol) * KSTR;
    const _Float16* aB1 = aB0 + (size_t)16 * KSTR;

    v8f az0 = {}, ah0 = {}, az1 = {}, ah1 = {};
    // ---- software-pipelined A-fragments (load ks+1 while ks computes) ----
    HFrag a0c, a1c;
    a0c.h[0] = *(const v8h*)&aB0[lhi * 16];
    a0c.h[1] = *(const v8h*)&aB0[lhi * 16 + 8];
    a1c.h[0] = *(const v8h*)&aB1[lhi * 16];
    a1c.h[1] = *(const v8h*)&aB1[lhi * 16 + 8];
#pragma unroll
    for (int ks = 0; ks < 16; ++ks) {
      HFrag a0n, a1n;
      if (ks + 1 < 16) {
        const int k0n = (ks + 1) * 32;
        // A k order used here: lane<16 holds K {0..15}; lane>=16 holds {16..31} (matches B pack)
        a0n.h[0] = *(const v8h*)&aB0[k0n + lhi * 16];
        a0n.h[1] = *(const v8h*)&aB0[k0n + lhi * 16 + 8];
        a1n.h[0] = *(const v8h*)&aB1[k0n + lhi * 16];
        a1n.h[1] = *(const v8h*)&aB1[k0n + lhi * 16 + 8];
      }
      az0 = __builtin_amdgcn_wmma_f32_16x16x32_f16(false, a0c.v, false, fz[ks], (short)0, az0, false, false);
      ah0 = __builtin_amdgcn_wmma_f32_16x16x32_f16(false, a0c.v, false, fh[ks], (short)0, ah0, false, false);
      az1 = __builtin_amdgcn_wmma_f32_16x16x32_f16(false, a1c.v, false, fz[ks], (short)0, az1, false, false);
      ah1 = __builtin_amdgcn_wmma_f32_16x16x32_f16(false, a1c.v, false, fh[ks], (short)0, ah1, false, false);
      if (ks + 1 < 16) {
        a0c = a0n;
        a1c = a1n;
      }
    }

    if (t + 1 < T_) stage(t + 1, nxt);                // overlap with gate math

    // ---- gate update (fp32): h = (1-z)*h + z*tanh(.) ; r-gate is dead code ----
#pragma unroll
    for (int r = 0; r < 8; ++r) {
      float z0 = 1.0f / (1.0f + __expf(-(az0[r] + czv)));
      float t0 = tanhf(ah0[r] + chv);
      h0[r] = (1.0f - z0) * h0[r] + z0 * t0;
      float z1 = 1.0f / (1.0f + __expf(-(az1[r] + czv)));
      float t1 = tanhf(ah1[r] + chv);
      h1[r] = (1.0f - z1) * h1[r] + z1 * t1;
    }
    __syncthreads();                                  // single barrier per step
  }

  // ---- spill final h (C layout: VGPR r -> row r + 8*lhi, col lane%16) ----
  const int gcol = ntglob * 16 + lcol;
#pragma unroll
  for (int r = 0; r < 8; ++r) {
    h_buf[(size_t)(b0 + mp * 32 + r + 8 * lhi) * H_ + gcol]      = h0[r];
    h_buf[(size_t)(b0 + mp * 32 + 16 + r + 8 * lhi) * H_ + gcol] = h1[r];
  }
}

// ---------------------------------------------------------------------------
// Output head: out[b] = sigmoid(h_last[b] . Wout + bout). Deterministic.
// ---------------------------------------------------------------------------
__global__ void grud_out_head(const float* __restrict__ h_buf,
                              const float* __restrict__ Wout,
                              const float* __restrict__ bout,
                              float* __restrict__ out) {
  int b = blockIdx.x * blockDim.x + threadIdx.x;
  if (b >= B_) return;
  float s = bout[0];
  const float* hr = h_buf + (size_t)b * H_;
  for (int j = 0; j < H_; ++j) s += hr[j] * Wout[j];
  out[b] = 1.0f / (1.0f + __expf(-s));
}

extern "C" void kernel_launch(void* const* d_in, const int* in_sizes, int n_in,
                              void* d_out, int out_size, void* d_ws, size_t ws_size,
                              hipStream_t stream) {
  const float* X     = (const float*)d_in[0];
  const float* M     = (const float*)d_in[1];
  const float* xm    = (const float*)d_in[2];
  const float* gamma = (const float*)d_in[3];
  const float* Wz    = (const float*)d_in[4];
  const float* bz    = (const float*)d_in[5];
  // d_in[6]=Wr, d_in[7]=br : dead code in reference (r_t unused)
  const float* Wh    = (const float*)d_in[8];
  const float* bh    = (const float*)d_in[9];
  const float* Wout  = (const float*)d_in[10];
  const float* bout  = (const float*)d_in[11];
  float* out = (float*)d_out;

  // Workspace: Wf (2MB, fragment-packed f16 weights) | cz(4KB) | ch(4KB) | h_buf(2MB)
  char* ws = (char*)d_ws;
  _Float16* Wf = (_Float16*)ws;
  float* cz    = (float*)(ws + (size_t)2 * NT_ * 16 * 512 * sizeof(_Float16));
  float* ch    = cz + H_;
  float* h_buf = ch + H_;

  const int nWf = 2 * NT_ * 16 * 512;
  grud_pack_weights<<<(nWf + 255) / 256, 256, 0, stream>>>(Wz, Wh, Wf);
  grud_bias_const<<<(H_ + 255) / 256, 256, 0, stream>>>(Wz, bz, Wh, bh, xm, cz, ch);

  size_t smem = (size_t)2 * BT * KSTR * sizeof(_Float16)   // double-buffered A tile
              + (size_t)2 * D_ * sizeof(float);            // gamma, xm  (~134 KB)
  dim3 grid(B_ / BT, H_ / HT);                             // 8 x 16 = 128 workgroups
  grud_main<<<grid, 256, smem, stream>>>(X, M, xm, gamma, Wf, cz, ch, h_buf);

  grud_out_head<<<(B_ + 255) / 256, 256, 0, stream>>>(h_buf, Wout, bout, out);
}
